// TemplatePairStackBlock_52183852646484
// MI455X (gfx1250) — matrix operand order repaired and checked by reading, not compiled, not run on hardware
//
#include <hip/hip_runtime.h>

// ---------------------------------------------------------------------------
// CDNA5 (gfx1250) TemplatePairStack block.
// All GEMMs run on v_wmma_f32_16x16x32_bf16 (bf16 operands, f32 accum).
// wave32; 256-thread (8-wave) workgroups; LDS-staged operands; TDM tile loads
// for the triangle GEMM where the toolchain exposes the builtin.
// ---------------------------------------------------------------------------

typedef __bf16 bf16_t;
typedef __attribute__((ext_vector_type(16))) __bf16 v16bf;
typedef __attribute__((ext_vector_type(8)))  float  v8f;
typedef __attribute__((ext_vector_type(4)))  unsigned int v4u;
typedef __attribute__((ext_vector_type(8)))  int    v8i;
typedef __attribute__((ext_vector_type(4)))  int    v4i;

#ifndef __has_builtin
#define __has_builtin(x) 0
#endif
#if __has_builtin(__builtin_amdgcn_tensor_load_to_lds) && \
    __has_builtin(__builtin_amdgcn_s_wait_tensorcnt)
#define HAVE_TDM 1
#else
#define HAVE_TDM 0
#endif

__device__ __forceinline__ v8f wmma_bf16(v16bf a, v16bf b, v8f c) {
  // D = A(16x32) * B(32x16) + C(16x16), f32 accumulate
  return __builtin_amdgcn_wmma_f32_16x16x32_bf16(false, a, false, b,
                                                 (short)0, c, false, false);
}

__device__ __forceinline__ v8f zero8() {
  v8f z = {0.f, 0.f, 0.f, 0.f, 0.f, 0.f, 0.f, 0.f};
  return z;
}

// A-operand (16x32 bf16) from row-major (rows x K) matrix, leading dim ld.
// Layout: lanes 0-15 hold M=lane, K = 0..15; lanes 16-31 hold
// M=lane-16, K = 16..31 (contiguous K, matches B loaders).
__device__ __forceinline__ v16bf load_a16(const bf16_t* base, int row0, int k0, int ld) {
  const int lane = threadIdx.x & 31;
  const int m = lane & 15;
  const int kb = (lane >> 4) * 16;
  const bf16_t* p = base + (size_t)(row0 + m) * ld + k0;
  v16bf a;
#pragma unroll
  for (int e = 0; e < 8; ++e) a[e] = p[kb + e];
#pragma unroll
  for (int e = 0; e < 8; ++e) a[8 + e] = p[kb + 8 + e];
  return a;
}

// B-operand (32x16 bf16) when B is stored TRANSPOSED (N x K row-major, ld):
// lane holds column n = lane&15, K contiguous (lanes 16-31 hold K+16 half).
__device__ __forceinline__ v16bf load_bt16(const bf16_t* base, int n0, int k0, int ld) {
  const int lane = threadIdx.x & 31;
  const int n = lane & 15;
  const int kb = (lane >> 4) * 16;
  const bf16_t* p = base + (size_t)(n0 + n) * ld + k0 + kb;
  v16bf b;
#pragma unroll
  for (int e = 0; e < 16; ++e) b[e] = p[e];
  return b;
}

// B-operand when B is stored K-major (K x N row-major, ld): strided reads.
__device__ __forceinline__ v16bf load_bk16(const bf16_t* base, int k0, int n0, int ld) {
  const int lane = threadIdx.x & 31;
  const int n = lane & 15;
  const int kb = (lane >> 4) * 16;
  const bf16_t* p = base + (size_t)(k0 + kb) * ld + n0 + n;
  v16bf b;
#pragma unroll
  for (int e = 0; e < 16; ++e) b[e] = p[(size_t)e * ld];
  return b;
}

__device__ __forceinline__ float sigmoidf(float x) {
  return 1.f / (1.f + __expf(-x));
}

// LayerNorm of one 64-channel token -> bf16 (b128 global loads)
__device__ __forceinline__ void layernorm64(const float* __restrict__ src,
                                            const float* __restrict__ s,
                                            const float* __restrict__ b,
                                            bf16_t* __restrict__ dst) {
  float buf[64];
  float mean = 0.f;
#pragma unroll
  for (int c4 = 0; c4 < 16; ++c4) {
    const float4 v = ((const float4*)src)[c4];
    buf[4 * c4 + 0] = v.x; buf[4 * c4 + 1] = v.y;
    buf[4 * c4 + 2] = v.z; buf[4 * c4 + 3] = v.w;
    mean += v.x + v.y + v.z + v.w;
  }
  mean *= 0.015625f;
  float var = 0.f;
#pragma unroll
  for (int c = 0; c < 64; ++c) { float d = buf[c] - mean; var += d * d; }
  var *= 0.015625f;
  const float inv = rsqrtf(var + 1e-5f);
#pragma unroll
  for (int c4 = 0; c4 < 16; ++c4) {
    const float4 sv = ((const float4*)s)[c4];
    const float4 bv = ((const float4*)b)[c4];
    dst[4 * c4 + 0] = (bf16_t)((buf[4 * c4 + 0] - mean) * inv * sv.x + bv.x);
    dst[4 * c4 + 1] = (bf16_t)((buf[4 * c4 + 1] - mean) * inv * sv.y + bv.y);
    dst[4 * c4 + 2] = (bf16_t)((buf[4 * c4 + 2] - mean) * inv * sv.z + bv.z);
    dst[4 * c4 + 3] = (bf16_t)((buf[4 * c4 + 3] - mean) * inv * sv.w + bv.w);
  }
}

#if HAVE_TDM
// TDM: DMA a 64-row x 32-col bf16 tile (row stride `stride_e` elements) from
// global memory into contiguous LDS at byte offset `lds_off`.
__device__ __forceinline__ void tdm_load_tile_64x32_bf16(
    const bf16_t* gsrc, unsigned lds_off, unsigned stride_e) {
  const unsigned long long ga = (unsigned long long)(size_t)gsrc;
  v4u g0;
  g0[0] = 1u;                                         // count=1, user D#
  g0[1] = lds_off;                                    // lds_addr (bytes)
  g0[2] = (unsigned)(ga & 0xffffffffu);               // global_addr[31:0]
  g0[3] = (unsigned)((ga >> 32) & 0x01ffffffu) | (2u << 30);  // [56:32]+type=2
  v8i g1;
  g1[0] = (int)(1u << 16);         // data_size = 2 bytes; no multicast
  g1[1] = (int)(256u << 16);       // tensor_dim0 = 256 (bits [63:48])
  g1[2] = (int)(256u << 16);       // tensor_dim1 = 256 (bits [95:80])
  g1[3] = (int)(32u << 16);        // tile_dim0  = 32  (bits [127:112])
  g1[4] = 64;                      // tile_dim1  = 64  (bits [143:128])
  g1[5] = (int)stride_e;           // tensor_dim0_stride[31:0]
  g1[6] = 0;                       // stride hi / dim1_stride lo
  g1[7] = 0;
  v4i z4 = {0, 0, 0, 0};
#if __clang_major__ >= 23
  v8i z8 = {0, 0, 0, 0, 0, 0, 0, 0};
  __builtin_amdgcn_tensor_load_to_lds(g0, g1, z4, z4, z8, 0);
#else
  __builtin_amdgcn_tensor_load_to_lds(g0, g1, z4, z4, 0);
#endif
}
#endif

// ---------------------------------------------------------------------------
// Kernel 1: fused gated triangle attention for one (row r, head h).
// Flash-style streaming softmax over 256 keys; Q/K/V/G staged in LDS (bf16).
// Writes o = g * softmax(q k^T / 8) v into obuf[(r,j,h*64+c)] (bf16).
// The reference's pair bias (z@Wb) is constant over the softmax axis -> no-op.
// ---------------------------------------------------------------------------
__global__ __launch_bounds__(256) void tri_attn_kernel(
    const float* __restrict__ tin,
    const float* __restrict__ ln_s, const float* __restrict__ ln_b,
    const float* __restrict__ Wq, const float* __restrict__ Wk,
    const float* __restrict__ Wv, const float* __restrict__ Wg,
    const float* __restrict__ bg,
    bf16_t* __restrict__ obuf, int transposed) {
  extern __shared__ __align__(16) char smem_raw[];
  bf16_t* zln = (bf16_t*)smem_raw;       // 256x64
  bf16_t* qs  = zln + 256 * 64;          // 256x64
  bf16_t* ks  = qs + 256 * 64;           // 256x64
  bf16_t* vs  = ks + 256 * 64;           // 256x64
  bf16_t* gs  = vs + 256 * 64;           // 256x64
  bf16_t* wt  = gs + 256 * 64;           // 64x64 (N x K)
  bf16_t* psA = wt + 64 * 64;            // 8 waves x (16x32) P staging
  float*  bvec = (float*)(psA + 8 * 16 * 32);  // 64

  const int tid = threadIdx.x;
  const int r = blockIdx.x >> 2;
  const int h = blockIdx.x & 3;
  const int wave = tid >> 5;
  const int lane = tid & 31;
  const int colX = lane & 15;
  const int rb = (lane >> 4) * 8;

  // ---- Phase 0: LayerNorm the 256 tokens of this row
  {
    const int j = tid;
    const float* zrow =
        tin + (size_t)(transposed ? (j * 256 + r) : (r * 256 + j)) * 64;
    layernorm64(zrow, ln_s, ln_b, zln + j * 64);
  }
  __syncthreads();

  // ---- Phase 1: q,k,v,g = zln @ W_head (64x64), g gets sigmoid(.+bg)
  const float* Wp[4] = {Wq, Wk, Wv, Wg};
  bf16_t* dstp[4] = {qs, ks, vs, gs};
#pragma unroll
  for (int p = 0; p < 4; ++p) {
    const float* W = Wp[p];
    // transpose W head-slice into wt[n][k], b128 global loads over n
    for (int idx = tid; idx < 1024; idx += 256) {
      const int kk = idx >> 4, n4 = (idx & 15) * 4;
      const float4 wv = *(const float4*)(W + kk * 256 + h * 64 + n4);
      wt[(n4 + 0) * 64 + kk] = (bf16_t)wv.x;
      wt[(n4 + 1) * 64 + kk] = (bf16_t)wv.y;
      wt[(n4 + 2) * 64 + kk] = (bf16_t)wv.z;
      wt[(n4 + 3) * 64 + kk] = (bf16_t)wv.w;
    }
    if (tid < 64) bvec[tid] = (p == 3) ? bg[h * 64 + tid] : 0.f;
    __syncthreads();
    bf16_t* dst = dstp[p];
    for (int id = wave; id < 64; id += 8) {
      const int mt = id >> 2, nt = id & 3;
      v8f acc = zero8();
#pragma unroll
      for (int k0 = 0; k0 < 64; k0 += 32)
        acc = wmma_bf16(load_a16(zln, mt * 16, k0, 64),
                        load_bt16(wt, nt * 16, k0, 64), acc);
      const float bias = bvec[nt * 16 + colX];
#pragma unroll
      for (int rr = 0; rr < 8; ++rr) {
        float x = acc[rr] + bias;
        if (p == 3) x = sigmoidf(x);
        dst[(mt * 16 + rb + rr) * 64 + nt * 16 + colX] = (bf16_t)x;
      }
    }
    __syncthreads();
  }

  // ---- Phase 2: flash attention. Wave owns 32 queries (2 m-tiles).
  const int q0 = wave * 32;
  bf16_t* ps = psA + wave * (16 * 32);
  v8f o[2][4];
  float M[2][8], L[2][8];
#pragma unroll
  for (int mt = 0; mt < 2; ++mt) {
#pragma unroll
    for (int nt = 0; nt < 4; ++nt) o[mt][nt] = zero8();
#pragma unroll
    for (int rr = 0; rr < 8; ++rr) { M[mt][rr] = -1e30f; L[mt][rr] = 0.f; }
  }

  for (int kt = 0; kt < 256; kt += 32) {
#pragma unroll
    for (int mt = 0; mt < 2; ++mt) {
      v8f s0 = zero8(), s1 = zero8();
#pragma unroll
      for (int k0 = 0; k0 < 64; k0 += 32) {
        v16bf a = load_a16(qs, q0 + mt * 16, k0, 64);
        s0 = wmma_bf16(a, load_bt16(ks, kt, k0, 64), s0);
        s1 = wmma_bf16(a, load_bt16(ks, kt + 16, k0, 64), s1);
      }
      float rmax[8], rsum[8];
#pragma unroll
      for (int rr = 0; rr < 8; ++rr) {
        s0[rr] *= 0.125f;  // C^-0.5
        s1[rr] *= 0.125f;
        rmax[rr] = fmaxf(s0[rr], s1[rr]);
      }
      // row-wise max: reduce over the 16 lanes of this half (row = rb+rr)
#pragma unroll
      for (int d = 1; d < 16; d <<= 1)
#pragma unroll
        for (int rr = 0; rr < 8; ++rr)
          rmax[rr] = fmaxf(rmax[rr], __shfl_xor(rmax[rr], d, 32));
#pragma unroll
      for (int rr = 0; rr < 8; ++rr) {
        const float mnew = fmaxf(M[mt][rr], rmax[rr]);
        const float alpha = __expf(M[mt][rr] - mnew);
        M[mt][rr] = mnew;
        s0[rr] = __expf(s0[rr] - mnew);
        s1[rr] = __expf(s1[rr] - mnew);
        rsum[rr] = s0[rr] + s1[rr];
        L[mt][rr] *= alpha;
#pragma unroll
        for (int nt = 0; nt < 4; ++nt) o[mt][nt][rr] *= alpha;
      }
#pragma unroll
      for (int d = 1; d < 16; d <<= 1)
#pragma unroll
        for (int rr = 0; rr < 8; ++rr)
          rsum[rr] += __shfl_xor(rsum[rr], d, 32);
#pragma unroll
      for (int rr = 0; rr < 8; ++rr) L[mt][rr] += rsum[rr];
      // Stage P (bf16, row-major 16x32) then consume as A-operand.
      // Same-wave LDS is in-order -> no barrier needed.
#pragma unroll
      for (int rr = 0; rr < 8; ++rr) {
        ps[(rb + rr) * 32 + colX] = (bf16_t)s0[rr];
        ps[(rb + rr) * 32 + 16 + colX] = (bf16_t)s1[rr];
      }
      v16bf pa = load_a16(ps, 0, 0, 32);
#pragma unroll
      for (int nt = 0; nt < 4; ++nt)
        o[mt][nt] = wmma_bf16(pa, load_bk16(vs, kt, nt * 16, 64), o[mt][nt]);
    }
  }

  // ---- finalize: o/L * g -> obuf
#pragma unroll
  for (int mt = 0; mt < 2; ++mt)
#pragma unroll
    for (int nt = 0; nt < 4; ++nt)
#pragma unroll
      for (int rr = 0; rr < 8; ++rr) {
        const int row = q0 + mt * 16 + rb + rr;
        const int col = nt * 16 + colX;
        float val = o[mt][nt][rr] / L[mt][rr];
        val *= (float)gs[row * 64 + col];
        obuf[((size_t)r * 256 + row) * 256 + h * 64 + col] = (bf16_t)val;
      }
}

// ---------------------------------------------------------------------------
// Kernel 2: out = resid + hin(65536x256) @ W(256x64) + bias, with optional
// i<->j remap (used for attn_end) applied to resid/out indices.
// ---------------------------------------------------------------------------
__global__ __launch_bounds__(256) void out_proj_kernel(
    const bf16_t* __restrict__ hin, const float* __restrict__ W,
    const float* __restrict__ bias, const float* __restrict__ resid,
    float* __restrict__ outp, int transposed) {
  extern __shared__ __align__(16) char smem_raw[];
  bf16_t* abuf = (bf16_t*)smem_raw;      // 64x256
  bf16_t* wt = abuf + 64 * 256;          // 64x256 (N x K)
  float* bvec = (float*)(wt + 64 * 256); // 64
  const int tid = threadIdx.x;
  const int tok0 = blockIdx.x * 64;
  // stage activations: 16-B (8 x bf16) vector copies
  {
    const uint4* src = (const uint4*)(hin + (size_t)tok0 * 256);
    uint4* dstv = (uint4*)abuf;
    for (int idx = tid; idx < 2048; idx += 256) dstv[idx] = src[idx];
  }
  // transpose W into wt[n][k], b128 global loads over n
  for (int idx = tid; idx < 4096; idx += 256) {
    const int k = idx >> 4, n4 = (idx & 15) * 4;
    const float4 wv = *(const float4*)(W + k * 64 + n4);
    wt[(n4 + 0) * 256 + k] = (bf16_t)wv.x;
    wt[(n4 + 1) * 256 + k] = (bf16_t)wv.y;
    wt[(n4 + 2) * 256 + k] = (bf16_t)wv.z;
    wt[(n4 + 3) * 256 + k] = (bf16_t)wv.w;
  }
  if (tid < 64) bvec[tid] = bias[tid];
  __syncthreads();
  const int wave = tid >> 5, lane = tid & 31;
  const int colX = lane & 15, rb = (lane >> 4) * 8;
  for (int id = wave; id < 16; id += 8) {
    const int mt = id >> 2, nt = id & 3;
    v8f acc = zero8();
#pragma unroll
    for (int k0 = 0; k0 < 256; k0 += 32)
      acc = wmma_bf16(load_a16(abuf, mt * 16, k0, 256),
                      load_bt16(wt, nt * 16, k0, 256), acc);
#pragma unroll
    for (int rr = 0; rr < 8; ++rr) {
      const int token = tok0 + mt * 16 + rb + rr;
      const int mapped =
          transposed ? ((token & 255) * 256 + (token >> 8)) : token;
      const int col = nt * 16 + colX;
      float v = acc[rr] + bvec[col];
      if (resid) v += resid[(size_t)mapped * 64 + col];
      outp[(size_t)mapped * 64 + col] = v;
    }
  }
}

// ---------------------------------------------------------------------------
// Kernel 3: triangle-mult projections. a=(z@Wa+ba)*sig(z@Wga+bga), likewise b,
// g=sig(z@Wgt+bgt). a/b scattered into (c, kept-dim, contract-dim) layout so
// the triangle einsum becomes 64 per-channel A @ B^T GEMMs.
// ---------------------------------------------------------------------------
__global__ __launch_bounds__(256) void mult_proj_kernel(
    const float* __restrict__ tin, const float* __restrict__ ln_s,
    const float* __restrict__ ln_b, const float* __restrict__ Wa,
    const float* __restrict__ ba, const float* __restrict__ Wga,
    const float* __restrict__ bga, const float* __restrict__ Wbp,
    const float* __restrict__ bbp, const float* __restrict__ Wgb,
    const float* __restrict__ bgb, const float* __restrict__ Wgt,
    const float* __restrict__ bgt, bf16_t* __restrict__ at,
    bf16_t* __restrict__ bt, bf16_t* __restrict__ gbuf, int outgoing) {
  extern __shared__ __align__(16) char smem_raw[];
  bf16_t* zln = (bf16_t*)smem_raw;     // 64x64
  bf16_t* wt = zln + 4096;             // 64x64 (N x K)
  float* bvec = (float*)(wt + 4096);   // 64
  const int tid = threadIdx.x;
  const int tok0 = blockIdx.x * 64;
  if (tid < 64)
    layernorm64(tin + (size_t)(tok0 + tid) * 64, ln_s, ln_b, zln + tid * 64);
  __syncthreads();

  const int wave = tid >> 5, lane = tid & 31;
  const int colX = lane & 15, rb = (lane >> 4) * 8;
  const float* WA[3] = {Wa, Wbp, Wgt};
  const float* BA[3] = {ba, bbp, bgt};
  const float* WG[2] = {Wga, Wgb};
  const float* BG[2] = {bga, bgb};

#pragma unroll
  for (int pp = 0; pp < 3; ++pp) {
    // linear part: transpose W (64x64) into wt[n][k]
    for (int idx = tid; idx < 1024; idx += 256) {
      const int k = idx >> 4, n4 = (idx & 15) * 4;
      const float4 wv = *(const float4*)(WA[pp] + k * 64 + n4);
      wt[(n4 + 0) * 64 + k] = (bf16_t)wv.x;
      wt[(n4 + 1) * 64 + k] = (bf16_t)wv.y;
      wt[(n4 + 2) * 64 + k] = (bf16_t)wv.z;
      wt[(n4 + 3) * 64 + k] = (bf16_t)wv.w;
    }
    if (tid < 64) bvec[tid] = BA[pp][tid];
    __syncthreads();
    v8f lin[2];
#pragma unroll
    for (int s = 0; s < 2; ++s) {
      const int id = wave * 2 + s;
      const int mt = id >> 2, nt = id & 3;
      v8f acc = zero8();
#pragma unroll
      for (int k0 = 0; k0 < 64; k0 += 32)
        acc = wmma_bf16(load_a16(zln, mt * 16, k0, 64),
                        load_bt16(wt, nt * 16, k0, 64), acc);
      const float bias = bvec[nt * 16 + colX];
#pragma unroll
      for (int rr = 0; rr < 8; ++rr) acc[rr] += bias;
      lin[s] = acc;
    }
    __syncthreads();

    if (pp < 2) {
      // gate part, then combine + scatter to (c, d0, d1)
      for (int idx = tid; idx < 1024; idx += 256) {
        const int k = idx >> 4, n4 = (idx & 15) * 4;
        const float4 wv = *(const float4*)(WG[pp] + k * 64 + n4);
        wt[(n4 + 0) * 64 + k] = (bf16_t)wv.x;
        wt[(n4 + 1) * 64 + k] = (bf16_t)wv.y;
        wt[(n4 + 2) * 64 + k] = (bf16_t)wv.z;
        wt[(n4 + 3) * 64 + k] = (bf16_t)wv.w;
      }
      if (tid < 64) bvec[tid] = BG[pp][tid];
      __syncthreads();
      bf16_t* dst = (pp == 0) ? at : bt;
#pragma unroll
      for (int s = 0; s < 2; ++s) {
        const int id = wave * 2 + s;
        const int mt = id >> 2, nt = id & 3;
        v8f acc = zero8();
#pragma unroll
        for (int k0 = 0; k0 < 64; k0 += 32)
          acc = wmma_bf16(load_a16(zln, mt * 16, k0, 64),
                          load_bt16(wt, nt * 16, k0, 64), acc);
        const float bias = bvec[nt * 16 + colX];
#pragma unroll
        for (int rr = 0; rr < 8; ++rr) {
          const int token = tok0 + mt * 16 + rb + rr;
          const int c = nt * 16 + colX;
          const float val = lin[s][rr] * sigmoidf(acc[rr] + bias);
          const int trow = token >> 8, tcol = token & 255;
          const int d0 = outgoing ? trow : tcol;   // kept dim
          const int d1 = outgoing ? tcol : trow;   // contraction dim
          dst[((size_t)c * 256 + d0) * 256 + d1] = (bf16_t)val;
        }
      }
      __syncthreads();
    } else {
      // g = sigmoid(lin) in natural (token, c) layout
#pragma unroll
      for (int s = 0; s < 2; ++s) {
        const int id = wave * 2 + s;
        const int mt = id >> 2, nt = id & 3;
#pragma unroll
        for (int rr = 0; rr < 8; ++rr) {
          const int token = tok0 + mt * 16 + rb + rr;
          const int c = nt * 16 + colX;
          gbuf[(size_t)token * 64 + c] = (bf16_t)sigmoidf(lin[s][rr]);
        }
      }
    }
  }
}

// ---------------------------------------------------------------------------
// Kernel 4: per-channel triangle GEMM x_c = A_c(256x256) @ B_c(256x256)^T,
// scattered back to (i,j,c) f32. Tiles staged by the Tensor Data Mover when
// available (64x32 bf16 tile, dim0_stride 256), else vectorized LDS copies.
// ---------------------------------------------------------------------------
__global__ __launch_bounds__(256) void tri_gemm_kernel(
    const bf16_t* __restrict__ at, const bf16_t* __restrict__ bt,
    float* __restrict__ xbuf) {
  extern __shared__ __align__(16) char smem_raw[];
  bf16_t* As = (bf16_t*)smem_raw;  // 64x32
  bf16_t* Bs = As + 2048;          // 64x32
  const int tid = threadIdx.x;
  const int c = blockIdx.x >> 4;
  const int ij = blockIdx.x & 15;
  const int i0 = (ij >> 2) * 64, j0 = (ij & 3) * 64;
  const bf16_t* A = at + (size_t)c * 65536;
  const bf16_t* B = bt + (size_t)c * 65536;
  const int wave = tid >> 5, lane = tid & 31;
  const int colX = lane & 15, rb = (lane >> 4) * 8;
#if HAVE_TDM
  const unsigned as_off = (unsigned)(size_t)(void*)As;
  const unsigned bs_off = (unsigned)(size_t)(void*)Bs;
#endif

  v8f acc[2] = {zero8(), zero8()};
  for (int k0 = 0; k0 < 256; k0 += 32) {
#if HAVE_TDM
    if (wave == 0) {
      tdm_load_tile_64x32_bf16(A + (size_t)i0 * 256 + k0, as_off, 256u);
      tdm_load_tile_64x32_bf16(B + (size_t)j0 * 256 + k0, bs_off, 256u);
      __builtin_amdgcn_s_wait_tensorcnt((short)0);
    }
#else
    {
      const uint4* A4 = (const uint4*)A;
      const uint4* B4 = (const uint4*)B;
      uint4* As4 = (uint4*)As;
      uint4* Bs4 = (uint4*)Bs;
      // 64 rows x 4 uint4 per row (32 bf16); one vector each per thread
      const int row = tid >> 2, chunk = tid & 3;
      As4[tid] = A4[(size_t)(i0 + row) * 32 + (k0 >> 3) + chunk];
      Bs4[tid] = B4[(size_t)(j0 + row) * 32 + (k0 >> 3) + chunk];
      if (k0 + 32 < 256) {
        __builtin_prefetch(A + (size_t)(i0 + (tid & 63)) * 256 + k0 + 32, 0, 1);
        __builtin_prefetch(B + (size_t)(j0 + (tid & 63)) * 256 + k0 + 32, 0, 1);
      }
    }
#endif
    __syncthreads();
#pragma unroll
    for (int s = 0; s < 2; ++s) {
      const int id = wave * 2 + s;
      const int mt = id >> 2, nt = id & 3;
      acc[s] = wmma_bf16(load_a16(As, mt * 16, 0, 32),
                         load_bt16(Bs, nt * 16, 0, 32), acc[s]);
    }
    __syncthreads();
  }
#pragma unroll
  for (int s = 0; s < 2; ++s) {
    const int id = wave * 2 + s;
    const int mt = id >> 2, nt = id & 3;
#pragma unroll
    for (int rr = 0; rr < 8; ++rr)
      xbuf[(((size_t)(i0 + mt * 16 + rb + rr)) * 256 + j0 + nt * 16 + colX) *
               64 + c] = acc[s][rr];
  }
}

// ---------------------------------------------------------------------------
// Kernel 5: triangle-mult output: out = g * (LN_c(x) @ Wo + bo). No residual.
// ---------------------------------------------------------------------------
__global__ __launch_bounds__(256) void mult_out_kernel(
    const float* __restrict__ xbuf, const float* __restrict__ ln_s,
    const float* __restrict__ ln_b, const float* __restrict__ Wo,
    const float* __restrict__ bo, const bf16_t* __restrict__ gbuf,
    float* __restrict__ outp) {
  extern __shared__ __align__(16) char smem_raw[];
  bf16_t* zln = (bf16_t*)smem_raw;   // 64x64
  bf16_t* wt = zln + 4096;           // 64x64
  float* bvec = (float*)(wt + 4096); // 64
  const int tid = threadIdx.x;
  const int tok0 = blockIdx.x * 64;
  if (tid < 64)
    layernorm64(xbuf + (size_t)(tok0 + tid) * 64, ln_s, ln_b, zln + tid * 64);
  for (int idx = tid; idx < 1024; idx += 256) {
    const int k = idx >> 4, n4 = (idx & 15) * 4;
    const float4 wv = *(const float4*)(Wo + k * 64 + n4);
    wt[(n4 + 0) * 64 + k] = (bf16_t)wv.x;
    wt[(n4 + 1) * 64 + k] = (bf16_t)wv.y;
    wt[(n4 + 2) * 64 + k] = (bf16_t)wv.z;
    wt[(n4 + 3) * 64 + k] = (bf16_t)wv.w;
  }
  if (tid < 64) bvec[tid] = bo[tid];
  __syncthreads();
  const int wave = tid >> 5, lane = tid & 31;
  const int colX = lane & 15, rb = (lane >> 4) * 8;
#pragma unroll
  for (int s = 0; s < 2; ++s) {
    const int id = wave * 2 + s;
    const int mt = id >> 2, nt = id & 3;
    v8f acc = zero8();
#pragma unroll
    for (int k0 = 0; k0 < 64; k0 += 32)
      acc = wmma_bf16(load_a16(zln, mt * 16, k0, 64),
                      load_bt16(wt, nt * 16, k0, 64), acc);
    const float bias = bvec[nt * 16 + colX];
#pragma unroll
    for (int rr = 0; rr < 8; ++rr) {
      const int token = tok0 + mt * 16 + rb + rr;
      const int col = nt * 16 + colX;
      const float g = (float)gbuf[(size_t)token * 64 + col];
      outp[(size_t)token * 64 + col] = g * (acc[rr] + bias);
    }
  }
}

// ---------------------------------------------------------------------------
// Kernel 6: transition hidden: h = relu(LN(t) @ Wi(64x256) + bi) -> bf16
// ---------------------------------------------------------------------------
__global__ __launch_bounds__(256) void trans_h_kernel(
    const float* __restrict__ tin, const float* __restrict__ ln_s,
    const float* __restrict__ ln_b, const float* __restrict__ Wi,
    const float* __restrict__ bi, bf16_t* __restrict__ hbuf) {
  extern __shared__ __align__(16) char smem_raw[];
  bf16_t* zln = (bf16_t*)smem_raw;      // 64x64
  bf16_t* wt = zln + 4096;              // 256x64 (N x K)
  float* bvec = (float*)(wt + 256 * 64);// 256
  const int tid = threadIdx.x;
  const int tok0 = blockIdx.x * 64;
  if (tid < 64)
    layernorm64(tin + (size_t)(tok0 + tid) * 64, ln_s, ln_b, zln + tid * 64);
  for (int idx = tid; idx < 4096; idx += 256) {
    const int k = idx >> 6, n4 = (idx & 63) * 4;
    const float4 wv = *(const float4*)(Wi + k * 256 + n4);
    wt[(n4 + 0) * 64 + k] = (bf16_t)wv.x;
    wt[(n4 + 1) * 64 + k] = (bf16_t)wv.y;
    wt[(n4 + 2) * 64 + k] = (bf16_t)wv.z;
    wt[(n4 + 3) * 64 + k] = (bf16_t)wv.w;
  }
  bvec[tid] = bi[tid];
  __syncthreads();
  const int wave = tid >> 5, lane = tid & 31;
  const int colX = lane & 15, rb = (lane >> 4) * 8;
  for (int id = wave; id < 64; id += 8) {
    const int mt = id >> 4, nt = id & 15;
    v8f acc = zero8();
#pragma unroll
    for (int k0 = 0; k0 < 64; k0 += 32)
      acc = wmma_bf16(load_a16(zln, mt * 16, k0, 64),
                      load_bt16(wt, nt * 16, k0, 64), acc);
    const float bias = bvec[nt * 16 + colX];
#pragma unroll
    for (int rr = 0; rr < 8; ++rr) {
      const int token = tok0 + mt * 16 + rb + rr;
      const float v = fmaxf(acc[rr] + bias, 0.f);
      hbuf[(size_t)token * 256 + nt * 16 + colX] = (bf16_t)v;
    }
  }
}

// ---------------------------------------------------------------------------
// Host side
// ---------------------------------------------------------------------------
struct AttnP { const float *ln_s,*ln_b,*Wq,*Wk,*Wv,*Wb,*Wg,*bg,*Wo,*bo; };
struct MultP { const float *ln_s,*ln_b,*ln_os,*ln_ob,*Wa,*ba,*Wga,*bga,
                           *Wbp,*bbp,*Wgb,*bgb,*Wgt,*bgt,*Wo,*bo; };
struct TransP { const float *ln_s,*ln_b,*Wi,*bi,*Wo,*bo; };
struct AllP { AttnP as, ae; MultP mo, mi; TransP tr; const float* t; };

static void fill_params(void* const* din, const int* sizes, int n, AllP& P) {
  int ti = n - 1;
  for (int i = 0; i < n; ++i)
    if (sizes[i] == 256 * 256 * 64) { ti = i; break; }
  auto F = [&](int i) { return (const float*)din[i]; };
  if (ti == 0) {
    // dict-insertion order: t, attn_start, attn_end, mult_out, mult_in, trans
    int b = 1;
    auto attn = [&](AttnP& a) {
      a.ln_s=F(b+0); a.ln_b=F(b+1); a.Wq=F(b+2); a.Wk=F(b+3); a.Wv=F(b+4);
      a.Wb=F(b+5); a.Wg=F(b+6); a.bg=F(b+7); a.Wo=F(b+8); a.bo=F(b+9); b+=10;
    };
    auto mult = [&](MultP& m) {
      m.ln_s=F(b+0); m.ln_b=F(b+1); m.ln_os=F(b+2); m.ln_ob=F(b+3);
      m.Wa=F(b+4); m.ba=F(b+5); m.Wga=F(b+6); m.bga=F(b+7);
      m.Wbp=F(b+8); m.bbp=F(b+9); m.Wgb=F(b+10); m.bgb=F(b+11);
      m.Wgt=F(b+12); m.bgt=F(b+13); m.Wo=F(b+14); m.bo=F(b+15); b+=16;
    };
    P.t = F(0);
    attn(P.as); attn(P.ae); mult(P.mo); mult(P.mi);
    P.tr.ln_s=F(b+0); P.tr.ln_b=F(b+1); P.tr.Wi=F(b+2); P.tr.bi=F(b+3);
    P.tr.Wo=F(b+4); P.tr.bo=F(b+5);
  } else {
    // JAX tree_flatten sorted-key order: params(attn_end, attn_start,
    // mult_in, mult_out, trans) first, then t.
    int b = 0;
    auto attn = [&](AttnP& a) {
      a.Wb=F(b+0); a.Wg=F(b+1); a.Wk=F(b+2); a.Wo=F(b+3); a.Wq=F(b+4);
      a.Wv=F(b+5); a.bg=F(b+6); a.bo=F(b+7); a.ln_b=F(b+8); a.ln_s=F(b+9);
      b+=10;
    };
    auto mult = [&](MultP& m) {
      m.Wa=F(b+0); m.Wbp=F(b+1); m.Wga=F(b+2); m.Wgb=F(b+3); m.Wgt=F(b+4);
      m.Wo=F(b+5); m.ba=F(b+6); m.bbp=F(b+7); m.bga=F(b+8); m.bgb=F(b+9);
      m.bgt=F(b+10); m.bo=F(b+11); m.ln_b=F(b+12); m.ln_ob=F(b+13);
      m.ln_os=F(b+14); m.ln_s=F(b+15); b+=16;
    };
    attn(P.ae); attn(P.as); mult(P.mi); mult(P.mo);
    P.tr.Wi=F(b+0); P.tr.Wo=F(b+1); P.tr.bi=F(b+2); P.tr.bo=F(b+3);
    P.tr.ln_b=F(b+4); P.tr.ln_s=F(b+5);
    P.t = F(ti);
  }
}

extern "C" void kernel_launch(void* const* d_in, const int* in_sizes, int n_in,
                              void* d_out, int out_size, void* d_ws,
                              size_t ws_size, hipStream_t stream) {
  (void)out_size; (void)ws_size;
  AllP P;
  fill_params(d_in, in_sizes, n_in, P);

  // Workspace carve (92.3 MB total):
  char* w = (char*)d_ws;
  float* tA = (float*)w;                             // 16.8 MB
  float* tB = (float*)(w + 16777216);                // 16.8 MB
  char* big = w + 33554432;                          // 33.5 MB union region
  bf16_t* obuf = (bf16_t*)big;                       // (256,256,256) bf16
  float*  xbuf = (float*)big;                        // (256,256,64)  f32
  bf16_t* hbuf = (bf16_t*)big;                       // (65536,256)   bf16
  bf16_t* a_t = (bf16_t*)(w + 67108864);             // (64,256,256)  bf16
  bf16_t* b_t = (bf16_t*)(w + 75497472);             // (64,256,256)  bf16
  bf16_t* g_  = (bf16_t*)(w + 83886080);             // (65536,64)    bf16

  constexpr int ATTN_LDS = 256*64*2*5 + 64*64*2 + 8*16*32*2 + 64*4;  // 180480
  constexpr int OUTP_LDS = 64*256*2*2 + 64*4;                        // 65792
  constexpr int MP_LDS   = 64*64*2*2 + 64*4;                         // 16640
  constexpr int TG_LDS   = 2*2048*2;                                 // 8192
  constexpr int TH_LDS   = 64*64*2 + 256*64*2 + 256*4;               // 41984

  const dim3 G(1024), B(256);

  // t1 = t + attn_start(t)
  tri_attn_kernel<<<G, B, ATTN_LDS, stream>>>(
      P.t, P.as.ln_s, P.as.ln_b, P.as.Wq, P.as.Wk, P.as.Wv, P.as.Wg, P.as.bg,
      obuf, 0);
  out_proj_kernel<<<G, B, OUTP_LDS, stream>>>(obuf, P.as.Wo, P.as.bo, P.t, tA, 0);

  // t2 = t1 + swap(attn_end(swap(t1)))
  tri_attn_kernel<<<G, B, ATTN_LDS, stream>>>(
      tA, P.ae.ln_s, P.ae.ln_b, P.ae.Wq, P.ae.Wk, P.ae.Wv, P.ae.Wg, P.ae.bg,
      obuf, 1);
  out_proj_kernel<<<G, B, OUTP_LDS, stream>>>(obuf, P.ae.Wo, P.ae.bo, tA, tB, 1);

  // t3 = mult_out(t2)   (outgoing; overwrites, no residual)
  mult_proj_kernel<<<G, B, MP_LDS, stream>>>(
      tB, P.mo.ln_s, P.mo.ln_b, P.mo.Wa, P.mo.ba, P.mo.Wga, P.mo.bga,
      P.mo.Wbp, P.mo.bbp, P.mo.Wgb, P.mo.bgb, P.mo.Wgt, P.mo.bgt,
      a_t, b_t, g_, 1);
  tri_gemm_kernel<<<G, B, TG_LDS, stream>>>(a_t, b_t, xbuf);
  mult_out_kernel<<<G, B, MP_LDS, stream>>>(
      xbuf, P.mo.ln_os, P.mo.ln_ob, P.mo.Wo, P.mo.bo, g_, tA);

  // t4 = mult_in(t3)    (incoming)
  mult_proj_kernel<<<G, B, MP_LDS, stream>>>(
      tA, P.mi.ln_s, P.mi.ln_b, P.mi.Wa, P.mi.ba, P.mi.Wga, P.mi.bga,
      P.mi.Wbp, P.mi.bbp, P.mi.Wgb, P.mi.bgb, P.mi.Wgt, P.mi.bgt,
      a_t, b_t, g_, 0);
  tri_gemm_kernel<<<G, B, TG_LDS, stream>>>(a_t, b_t, xbuf);
  mult_out_kernel<<<G, B, MP_LDS, stream>>>(
      xbuf, P.mi.ln_os, P.mi.ln_ob, P.mi.Wo, P.mi.bo, g_, tB);

  // out = t4 + transition(t4)
  trans_h_kernel<<<G, B, TH_LDS, stream>>>(
      tB, P.tr.ln_s, P.tr.ln_b, P.tr.Wi, P.tr.bi, hbuf);
  out_proj_kernel<<<G, B, OUTP_LDS, stream>>>(
      hbuf, P.tr.Wo, P.tr.bo, tB, (float*)d_out, 0);
}
